// Attention_35476429865089
// MI455X (gfx1250) — matrix operand order repaired and checked by reading, not compiled, not run on hardware
//
#include <hip/hip_runtime.h>
#include <hip/hip_bf16.h>

// ---------------------------------------------------------------------------
// MLA attention forward for MI455X (gfx1250): bf16 WMMA + TDM-staged LDS tiles.
// D_MODEL=2048, D_HEAD=128, N_HEADS=16, D_LATENT=512, D_ROPE=64, B=2, T=2048
// ---------------------------------------------------------------------------

typedef __bf16 bf16;
typedef __bf16 v8bf  __attribute__((ext_vector_type(8)));
typedef __bf16 v16bf __attribute__((ext_vector_type(16)));
typedef float  v8f   __attribute__((ext_vector_type(8)));
typedef unsigned int u32x4 __attribute__((ext_vector_type(4)));
typedef int          i32x8 __attribute__((ext_vector_type(8)));
typedef int          i32x4 __attribute__((ext_vector_type(4)));

#define D_MODEL 2048
#define D_HEAD  128
#define N_HEADS 16
#define D_ROPE  64
#define DQK     192   // D_HEAD + D_ROPE
#define MLA_SCALE 0.07216878364870322f  // 1/sqrt(192)

// LDS row padding done by the TDM: pad_interval = 32 DWORDs (code 4),
// pad_amount = 4 DWORDs (code 3)  ->  every 64 bf16 elements get +8 elements.
#define PAD_IV 4u
#define PAD_AM 3u
#define PADROW64  72   // 64-elem row  -> 72-elem LDS stride (144B, conflict-free)
#define PADROW192 216  // 192-elem row -> 216-elem LDS stride (432B)

__device__ inline v8f wmma_bf16(v16bf a, v16bf b, v8f c) {
  return __builtin_amdgcn_wmma_f32_16x16x32_bf16(false, a, false, b, (short)0, c, false, false);
}

__device__ inline v16bf frag_cat(v8bf lo, v8bf hi) {
  return __builtin_shufflevector(lo, hi, 0,1,2,3,4,5,6,7,8,9,10,11,12,13,14,15);
}

// A fragment, 16x32 (MxK): lanes 0-15: M=lane, K=0..15;
// lanes 16-31: M=lane-16, K=16..31
__device__ inline v16bf load_a_frag(const bf16* A, int lda, int lane) {
  int m  = lane & 15;
  int kh = (lane >> 4) << 4;
  const bf16* p = A + (size_t)m * lda + kh;
  v8bf lo = *(const v8bf*)p;
  v8bf hi = *(const v8bf*)(p + 8);
  return frag_cat(lo, hi);
}

// B fragment, 32x16 (KxN) from row-major (N,K): lanes 0-15: N=lane,K=0..15;
// lanes 16-31: N=lane-16,K=16..31
__device__ inline v16bf load_b_frag(const bf16* W, int ldw, int lane) {
  int n  = lane & 15;
  int kh = (lane >> 4) << 4;
  const bf16* p = W + (size_t)n * ldw + kh;
  v8bf lo = *(const v8bf*)p;
  v8bf hi = *(const v8bf*)(p + 8);
  return frag_cat(lo, hi);
}

// B fragment from a TDM-padded 192-elem row (chunked: 64 elems + 8 pad).
__device__ inline v16bf load_b_frag_kpad(const bf16* base, int row0, int c0, int lane) {
  int n  = lane & 15;
  int kh = (lane >> 4) << 4;
  int c  = c0 + kh;                        // 0..191, (c&63) in {0,16,32,48}
  int off = (c >> 6) * PADROW64 + (c & 63);
  const bf16* p = base + (size_t)(row0 + n) * PADROW192 + off;
  v8bf lo = *(const v8bf*)p;
  v8bf hi = *(const v8bf*)(p + 8);
  return frag_cat(lo, hi);
}

// ---------------------------------------------------------------------------
// Tensor Data Mover: 2D tile (global, row-major, bf16) -> LDS, with row pad.
// D# layout per CDNA5 ISA 8.3/8.4 (group0: count/lds/global/type; group1:
// data_size, pad, tensor dims/strides, tile dims).
// ---------------------------------------------------------------------------
__device__ inline void tdm_load_2d(const void* gaddr, unsigned lds_off,
                                   unsigned tensor_d0, unsigned tensor_d1,
                                   unsigned stride0,
                                   unsigned tile_d0, unsigned tile_d1) {
  unsigned long long ga = (unsigned long long)gaddr;
  u32x4 g0;
  g0[0] = 1u;                                   // count=1 (valid user descriptor)
  g0[1] = lds_off;                              // lds_addr (bytes)
  g0[2] = (unsigned)(ga & 0xFFFFFFFFu);         // global_addr[31:0]
  g0[3] = (unsigned)((ga >> 32) & 0x01FFFFFFu)  // global_addr[56:32]
        | (2u << 30);                           // type = 2 ("image")
  i32x8 g1;
  g1[0] = (int)((1u << 16)                      // data_size = 1 -> 2 bytes
              | (1u << 20)                      // pad_enable
              | (PAD_IV << 22) | (PAD_AM << 25));
  g1[1] = (int)((tensor_d0 & 0xFFFFu) << 16);   // abar_addr=0 | tensor_dim0[15:0]
  g1[2] = (int)(((tensor_d0 >> 16) & 0xFFFFu) | ((tensor_d1 & 0xFFFFu) << 16));
  g1[3] = (int)(((tensor_d1 >> 16) & 0xFFFFu) | ((tile_d0 & 0xFFFFu) << 16));
  g1[4] = (int)(tile_d1 & 0xFFFFu);             // tile_dim1 | tile_dim2=0
  g1[5] = (int)stride0;                         // tensor_dim0_stride[31:0]
  g1[6] = 0;                                    // stride0[47:32] | stride1[15:0]
  g1[7] = 0;
  i32x4 gz  = {0, 0, 0, 0};
  i32x8 gz8 = {0, 0, 0, 0, 0, 0, 0, 0};
  // amdgpu-toolchain (clang-23) 6-arg form: (g0, g1, g2, g3, g4, cpol)
  __builtin_amdgcn_tensor_load_to_lds(g0, g1, gz, gz, gz8, 0);
}

__device__ inline unsigned lds_off_of(const void* p) {
  // generic shared address: high 32 = aperture, low 32 = LDS byte offset
  return (unsigned)(unsigned long long)p;
}

// ---------------------------------------------------------------------------
// fp32 -> bf16 conversion
// ---------------------------------------------------------------------------
__global__ void f32_to_bf16_k(const float* __restrict__ s, bf16* __restrict__ d, size_t n) {
  size_t i = (size_t)blockIdx.x * blockDim.x + threadIdx.x;
  size_t st = (size_t)gridDim.x * blockDim.x;
  for (; i < n; i += st) d[i] = (bf16)s[i];
}

// ---------------------------------------------------------------------------
// C(MxN) = A(MxK,bf16) @ W(NxK,bf16)^T, TDM double-buffered through LDS.
// block: 256 threads = 8 waves as 2(M) x 4(N); wave tile 64x32; block 128x128.
// K stepped by 64 (two 32-wide WMMA sub-steps). M % 128 == 0; K % 64 == 0.
// N handled by TDM zero-fill + store guard.
// ---------------------------------------------------------------------------
template<bool OUT_BF16>
__global__ __launch_bounds__(256) void gemm_bf16_wmma(
    const bf16* __restrict__ A, const bf16* __restrict__ W, void* __restrict__ Cv,
    int M, int N, int K) {
  const int lane = threadIdx.x & 31;
  const int wave = threadIdx.x >> 5;
  const int wm = wave >> 2, wn = wave & 3;
  const int mbase = blockIdx.y * 128 + wm * 64;
  const int nbase = blockIdx.x * 128 + wn * 32;

  __shared__ bf16 As[2][128 * PADROW64];
  __shared__ bf16 Bs[2][128 * PADROW64];

  const int wv = __builtin_amdgcn_readfirstlane(wave);  // scalar branch for TDM

  const bf16* Ablk = A + (size_t)(blockIdx.y * 128) * K;
  const bf16* Wblk = W + (size_t)(blockIdx.x * 128) * K;

  // prologue: stage k0 = 0 tiles
  if (wv == 0) tdm_load_2d(Ablk, lds_off_of(&As[0][0]), K, M, K, 64, 128);
  if (wv == 1) tdm_load_2d(Wblk, lds_off_of(&Bs[0][0]), K, N, K, 64, 128);
  __builtin_amdgcn_s_wait_tensorcnt(0);
  __syncthreads();

  const v8f vz = {0.f,0.f,0.f,0.f,0.f,0.f,0.f,0.f};
  v8f acc[4][2] = {{vz,vz},{vz,vz},{vz,vz},{vz,vz}};

  int cur = 0;
  for (int k0 = 0; k0 < K; k0 += 64) {
    if (k0 + 64 < K) {   // stage next tile into the other buffer
      if (wv == 0) tdm_load_2d(Ablk + (k0 + 64), lds_off_of(&As[cur ^ 1][0]), K, M, K, 64, 128);
      if (wv == 1) tdm_load_2d(Wblk + (k0 + 64), lds_off_of(&Bs[cur ^ 1][0]), K, N, K, 64, 128);
    }
#pragma unroll
    for (int kk = 0; kk < 64; kk += 32) {
      v16bf a[4], b[2];
#pragma unroll
      for (int i = 0; i < 4; ++i)
        a[i] = load_a_frag(&As[cur][0] + (size_t)(wm * 64 + i * 16) * PADROW64 + kk, PADROW64, lane);
#pragma unroll
      for (int j = 0; j < 2; ++j)
        b[j] = load_b_frag(&Bs[cur][0] + (size_t)(wn * 32 + j * 16) * PADROW64 + kk, PADROW64, lane);
#pragma unroll
      for (int i = 0; i < 4; ++i)
#pragma unroll
        for (int j = 0; j < 2; ++j)
          acc[i][j] = wmma_bf16(a[i], b[j], acc[i][j]);
    }
    __builtin_amdgcn_s_wait_tensorcnt(0);
    __syncthreads();
    cur ^= 1;
  }

  const int n = lane & 15, half = lane >> 4;
#pragma unroll
  for (int i = 0; i < 4; ++i)
#pragma unroll
    for (int j = 0; j < 2; ++j) {
      int col = nbase + j * 16 + n;
      if (col < N) {
#pragma unroll
        for (int r = 0; r < 8; ++r) {
          int row = mbase + i * 16 + half * 8 + r;
          float v = acc[i][j][r];
          if (OUT_BF16) ((bf16*)Cv)[(size_t)row * N + col] = (bf16)v;
          else          ((float*)Cv)[(size_t)row * N + col] = v;
        }
      }
    }
}

// ---------------------------------------------------------------------------
// RoPE + layout assembly (unchanged math).
// ---------------------------------------------------------------------------
__global__ __launch_bounds__(256) void rope_assemble_k(
    const bf16* __restrict__ qb, const float* __restrict__ qrot,
    const float* __restrict__ krot, const bf16* __restrict__ ukv,
    bf16* __restrict__ qf, bf16* __restrict__ kf, bf16* __restrict__ vt, int T) {
  const int bt = blockIdx.x;
  const int b = bt / T, t = bt - b * T;
  const int tid = threadIdx.x;

  __shared__ float ks[D_ROPE];
  if (tid < 32) {
    float inv = __powf(10000.0f, -(float)(2 * tid) / (float)D_ROPE);
    float ang = (float)t * inv;
    float c = __cosf(ang), sn = __sinf(ang);
    float x1 = krot[(size_t)bt * D_ROPE + tid];
    float x2 = krot[(size_t)bt * D_ROPE + tid + 32];
    ks[tid]      = x1 * c - x2 * sn;
    ks[tid + 32] = x2 * c + x1 * sn;
  }
  __syncthreads();

  for (int idx = tid; idx < N_HEADS * DQK; idx += 256) {
    int h = idx / DQK, d = idx - h * DQK;
    size_t dst = (((size_t)(b * N_HEADS + h)) * T + t) * DQK + d;
    float qv, kv;
    if (d < D_HEAD) {
      qv = (float)qb [(size_t)bt * D_MODEL       + h * D_HEAD + d];
      kv = (float)ukv[(size_t)bt * (2 * D_MODEL) + h * (2 * D_HEAD) + d];
    } else {
      int dr = d - D_HEAD;
      int i = dr & 31;
      float inv = __powf(10000.0f, -(float)(2 * i) / (float)D_ROPE);
      float ang = (float)t * inv;
      float c = __cosf(ang), sn = __sinf(ang);
      float x1 = qrot[(size_t)bt * (N_HEADS * D_ROPE) + h * D_ROPE + i];
      float x2 = qrot[(size_t)bt * (N_HEADS * D_ROPE) + h * D_ROPE + i + 32];
      qv = (dr < 32) ? (x1 * c - x2 * sn) : (x2 * c + x1 * sn);
      kv = ks[dr];
    }
    qf[dst] = (bf16)qv;
    kf[dst] = (bf16)kv;
  }

  for (int idx = tid; idx < N_HEADS * D_HEAD; idx += 256) {
    int h = idx / D_HEAD, d = idx - h * D_HEAD;
    vt[(((size_t)(b * N_HEADS + h)) * D_HEAD + d) * T + t] =
        ukv[(size_t)bt * (2 * D_MODEL) + h * (2 * D_HEAD) + D_HEAD + d];
  }
}

// ---------------------------------------------------------------------------
// Causal flash attention: TDM-staged K/V tiles, WMMA from LDS.
// grid = (T/64, B*H), block = 128 (4 waves); wave owns 16 query rows.
// All waves run the same block-uniform K loop (mask handles causality;
// k0 <= q0 always holds, so every tile has live keys for every wave).
// ---------------------------------------------------------------------------
__global__ __launch_bounds__(128) void mla_attn_k(
    const bf16* __restrict__ qf, const bf16* __restrict__ kf,
    const bf16* __restrict__ vt, bf16* __restrict__ ao, int T) {
  const int lane = threadIdx.x & 31;
  const int wave = threadIdx.x >> 5;
  const int bh = blockIdx.y;
  const int h = bh & (N_HEADS - 1), b = bh >> 4;
  const int q0 = blockIdx.x * 64 + wave * 16;

  const bf16* Q  = qf + ((size_t)bh * T + q0) * DQK;
  const bf16* Kp = kf + (size_t)bh * T * DQK;
  const bf16* Vp = vt + (size_t)bh * D_HEAD * T;

  __shared__ bf16 Ks[2][64 * PADROW192];   // 64 keys x 192 (padded rows)
  __shared__ bf16 Vs[2][128 * PADROW64];   // 128 d x 64 keys (padded rows)
  __shared__ bf16 Plds[4][16][64];

  const int wv = __builtin_amdgcn_readfirstlane(wave);

  // Preload Q fragments (16 x 192 = 6 A-fragments)
  v16bf qa[6];
#pragma unroll
  for (int d = 0; d < 6; ++d) qa[d] = load_a_frag(Q + d * 32, DQK, lane);

  // prologue: stage k-block 0
  if (wv == 0) tdm_load_2d(Kp, lds_off_of(&Ks[0][0]), DQK, T, DQK, DQK, 64);
  if (wv == 1) tdm_load_2d(Vp, lds_off_of(&Vs[0][0]), T, D_HEAD, T, 64, D_HEAD);
  __builtin_amdgcn_s_wait_tensorcnt(0);
  __syncthreads();

  const v8f vz = {0.f,0.f,0.f,0.f,0.f,0.f,0.f,0.f};
  v8f acc[8] = {vz,vz,vz,vz,vz,vz,vz,vz};
  float mrow[8], lrow[8];
#pragma unroll
  for (int r = 0; r < 8; ++r) { mrow[r] = -3.0e38f; lrow[r] = 0.f; }

  const int n = lane & 15, half = lane >> 4;
  const int kblocks = blockIdx.x + 1;

  int cur = 0;
  for (int kb = 0; kb < kblocks; ++kb) {
    const int k0 = kb * 64;
    if (kb + 1 < kblocks) {
      const int kn = k0 + 64;
      if (wv == 0) tdm_load_2d(Kp + (size_t)kn * DQK, lds_off_of(&Ks[cur ^ 1][0]), DQK, T, DQK, DQK, 64);
      if (wv == 1) tdm_load_2d(Vp + kn,               lds_off_of(&Vs[cur ^ 1][0]), T, D_HEAD, T, 64, D_HEAD);
    }

    // ---- S = Q K^T (16x64, fp32) from the LDS K tile ----
    v8f s[4] = {vz,vz,vz,vz};
#pragma unroll
    for (int d = 0; d < 6; ++d)
#pragma unroll
      for (int j = 0; j < 4; ++j) {
        v16bf kbf = load_b_frag_kpad(&Ks[cur][0], j * 16, d * 32, lane);
        s[j] = wmma_bf16(qa[d], kbf, s[j]);
      }

    // ---- scale + causal mask ----
    float sv[4][8];
#pragma unroll
    for (int j = 0; j < 4; ++j) {
      int key = k0 + j * 16 + n;
#pragma unroll
      for (int r = 0; r < 8; ++r) {
        int qrow = q0 + half * 8 + r;
        float v = s[j][r] * MLA_SCALE;
        sv[j][r] = (key > qrow) ? -1.0e30f : v;
      }
    }

    // ---- online softmax (row stats per 16-lane half) ----
#pragma unroll
    for (int r = 0; r < 8; ++r) {
      float mx = fmaxf(fmaxf(sv[0][r], sv[1][r]), fmaxf(sv[2][r], sv[3][r]));
#pragma unroll
      for (int off = 1; off < 16; off <<= 1)
        mx = fmaxf(mx, __shfl_xor(mx, off, 32));
      float mnew = fmaxf(mrow[r], mx);
      float sc = __expf(mrow[r] - mnew);
      mrow[r] = mnew;
      float rs = 0.f;
#pragma unroll
      for (int j = 0; j < 4; ++j) {
        sv[j][r] = __expf(sv[j][r] - mnew);
        rs += sv[j][r];
      }
#pragma unroll
      for (int off = 1; off < 16; off <<= 1)
        rs += __shfl_xor(rs, off, 32);
      lrow[r] = lrow[r] * sc + rs;
#pragma unroll
      for (int dt = 0; dt < 8; ++dt) acc[dt][r] = acc[dt][r] * sc;
    }

    // ---- stage P in LDS (same-wave LDS ops are in-order) ----
#pragma unroll
    for (int j = 0; j < 4; ++j)
#pragma unroll
      for (int r = 0; r < 8; ++r)
        Plds[wave][half * 8 + r][j * 16 + n] = (bf16)sv[j][r];

    // ---- O += P V from the LDS V tile ----
#pragma unroll
    for (int kk = 0; kk < 2; ++kk) {
      v16bf pa = load_a_frag(&Plds[wave][0][0] + kk * 32, 64, lane);
#pragma unroll
      for (int dt = 0; dt < 8; ++dt) {
        v16bf vb = load_b_frag(&Vs[cur][0] + (size_t)(dt * 16) * PADROW64 + kk * 32, PADROW64, lane);
        acc[dt] = wmma_bf16(pa, vb, acc[dt]);
      }
    }

    __builtin_amdgcn_s_wait_tensorcnt(0);
    __syncthreads();
    cur ^= 1;
  }

  // ---- normalize + write (B,T,H*128) bf16 ----
#pragma unroll
  for (int dt = 0; dt < 8; ++dt)
#pragma unroll
    for (int r = 0; r < 8; ++r) {
      float o = acc[dt][r] / lrow[r];
      size_t row = (size_t)b * T + q0 + half * 8 + r;
      ao[row * D_MODEL + h * D_HEAD + dt * 16 + n] = (bf16)o;
    }
}

// ---------------------------------------------------------------------------
// Host launcher
// ---------------------------------------------------------------------------
extern "C" void kernel_launch(void* const* d_in, const int* in_sizes, int n_in,
                              void* d_out, int out_size, void* d_ws, size_t ws_size,
                              hipStream_t stream) {
  (void)in_sizes; (void)n_in; (void)out_size; (void)ws_size;
  const int B = 2, T = 2048, M = B * T;     // M = 4096 rows

  const float* x      = (const float*)d_in[0];
  const float* w_q    = (const float*)d_in[1];
  const float* w_dkv  = (const float*)d_in[2];
  const float* w_ukv  = (const float*)d_in[3];
  const float* w_o    = (const float*)d_in[4];
  const float* w_qrot = (const float*)d_in[5];
  const float* w_krot = (const float*)d_in[6];
  // d_in[7] = mask (causal, recomputed in-kernel)

  char* base = (char*)d_ws;
  size_t off = 0;
  auto alloc = [&](size_t bytes) -> char* {
    char* p = base + off;
    off += (bytes + 65535) & ~(size_t)65535;
    return p;
  };

  bf16*  xb     = (bf16*)alloc((size_t)M * D_MODEL * 2);
  bf16*  wqb    = (bf16*)alloc((size_t)D_MODEL * D_MODEL * 2);
  bf16*  wdkvb  = (bf16*)alloc((size_t)512 * D_MODEL * 2);
  bf16*  wukvb  = (bf16*)alloc((size_t)(2 * D_MODEL) * 512 * 2);
  bf16*  wqrotb = (bf16*)alloc((size_t)(N_HEADS * D_ROPE) * D_MODEL * 2);
  bf16*  wkrotb = (bf16*)alloc((size_t)D_ROPE * D_MODEL * 2);
  bf16*  wob    = (bf16*)alloc((size_t)D_MODEL * D_MODEL * 2);
  bf16*  qb     = (bf16*)alloc((size_t)M * D_MODEL * 2);
  float* qrotf  = (float*)alloc((size_t)M * (N_HEADS * D_ROPE) * 4);
  float* krotf  = (float*)alloc((size_t)M * D_ROPE * 4);
  bf16*  dkvb   = (bf16*)alloc((size_t)M * 512 * 2);
  bf16*  ukvb   = (bf16*)alloc((size_t)M * (2 * D_MODEL) * 2);
  bf16*  qfull  = (bf16*)alloc((size_t)B * N_HEADS * T * DQK * 2);
  bf16*  kfull  = (bf16*)alloc((size_t)B * N_HEADS * T * DQK * 2);
  bf16*  vtb    = (bf16*)alloc((size_t)B * N_HEADS * D_HEAD * T * 2);
  bf16*  aob    = (bf16*)alloc((size_t)M * D_MODEL * 2);

  auto cvt = [&](const float* s, bf16* d, size_t n) {
    int grid = (int)((n + 255) / 256); if (grid > 2048) grid = 2048;
    f32_to_bf16_k<<<grid, 256, 0, stream>>>(s, d, n);
  };
  cvt(x,      xb,     (size_t)M * D_MODEL);
  cvt(w_q,    wqb,    (size_t)D_MODEL * D_MODEL);
  cvt(w_dkv,  wdkvb,  (size_t)512 * D_MODEL);
  cvt(w_ukv,  wukvb,  (size_t)(2 * D_MODEL) * 512);
  cvt(w_qrot, wqrotb, (size_t)(N_HEADS * D_ROPE) * D_MODEL);
  cvt(w_krot, wkrotb, (size_t)D_ROPE * D_MODEL);
  cvt(w_o,    wob,    (size_t)D_MODEL * D_MODEL);

  auto gemm_grid = [&](int N) { return dim3((unsigned)((N + 127) / 128), (unsigned)(M / 128)); };

  gemm_bf16_wmma<true ><<<gemm_grid(D_MODEL),          256, 0, stream>>>(xb,   wqb,    qb,    M, D_MODEL,          D_MODEL);
  gemm_bf16_wmma<true ><<<gemm_grid(512),              256, 0, stream>>>(xb,   wdkvb,  dkvb,  M, 512,              D_MODEL);
  gemm_bf16_wmma<false><<<gemm_grid(N_HEADS * D_ROPE), 256, 0, stream>>>(xb,   wqrotb, qrotf, M, N_HEADS * D_ROPE, D_MODEL);
  gemm_bf16_wmma<false><<<gemm_grid(D_ROPE),           256, 0, stream>>>(xb,   wkrotb, krotf, M, D_ROPE,           D_MODEL);
  gemm_bf16_wmma<true ><<<gemm_grid(2 * D_MODEL),      256, 0, stream>>>(dkvb, wukvb,  ukvb,  M, 2 * D_MODEL,      512);

  rope_assemble_k<<<M, 256, 0, stream>>>(qb, qrotf, krotf, ukvb, qfull, kfull, vtb, T);

  mla_attn_k<<<dim3(T / 64, B * N_HEADS), 128, 0, stream>>>(qfull, kfull, vtb, aob, T);

  gemm_bf16_wmma<false><<<gemm_grid(D_MODEL), 256, 0, stream>>>(aob, wob, (float*)d_out, M, D_MODEL, D_MODEL);
}
